// MultiheadAttention_75917841924285
// MI455X (gfx1250) — matrix operand order repaired and checked by reading, not compiled, not run on hardware
//
#include <hip/hip_runtime.h>
#include <hip/hip_bf16.h>
#include <math.h>

// ---------------------------------------------------------------------------
// Types for CDNA5 WMMA (wave32): v_wmma_f32_16x16x32_bf16
// ---------------------------------------------------------------------------
typedef __attribute__((ext_vector_type(16))) __bf16        v16bf;
typedef __attribute__((ext_vector_type(8)))  __bf16        v8bf;
typedef __attribute__((ext_vector_type(8)))  float         v8f;
typedef __attribute__((ext_vector_type(4)))  unsigned int  v4u;
typedef __attribute__((ext_vector_type(4)))  int           v4i;
typedef __attribute__((ext_vector_type(8)))  int           v8i_t;

// TDM available? (device pass, gfx1250, builtin present)
#if defined(__HIP_DEVICE_COMPILE__) && defined(__gfx1250__) && \
    __has_builtin(__builtin_amdgcn_tensor_load_to_lds)
#define USE_TDM 1
#else
#define USE_TDM 0
#endif

static __device__ inline v8f wmma_bf16(v16bf a, v16bf b, v8f c) {
    return __builtin_amdgcn_wmma_f32_16x16x32_bf16(
        false, a, false, b, (short)0, c, false, false);
}

// ---------------------------------------------------------------------------
// Fragment loaders per ISA 7.12.2 (wave32 layouts); work on global or LDS ptrs
// ---------------------------------------------------------------------------
static __device__ inline v16bf load_frag_a(const __bf16* base, int stride, int lane) {
    int r  = lane & 15;
    int kb = (lane < 16) ? 0 : 8;
    const __bf16* p = base + (size_t)r * stride + kb;
    union { v16bf v; v8bf h[2]; } u;
    u.h[0] = *(const v8bf*)(p);
    u.h[1] = *(const v8bf*)(p + 16);
    return u.v;
}

static __device__ inline v16bf load_frag_bT(const __bf16* base, int stride, int lane) {
    int n  = lane & 15;
    int kh = (lane < 16) ? 0 : 8;
    const __bf16* p = base + (size_t)n * stride + kh;
    union { v16bf v; v8bf h[2]; } u;
    u.h[0] = *(const v8bf*)(p);
    u.h[1] = *(const v8bf*)(p + 16);
    return u.v;
}

static __device__ inline v16bf load_frag_a_f32(const float* base, int stride, int lane) {
    int r  = lane & 15;
    int kb = (lane < 16) ? 0 : 8;
    const float* p = base + (size_t)r * stride + kb;
    v16bf v;
#pragma unroll
    for (int j = 0; j < 8; ++j) {
        v[j]     = (__bf16)p[j];
        v[8 + j] = (__bf16)p[16 + j];
    }
    return v;
}

#if USE_TDM
// Low 32 bits of a generic pointer into LDS == LDS byte offset (aperture is in [63:32])
static __device__ inline unsigned lds_off(const void* p) {
    return (unsigned)(uintptr_t)p;
}

// TDM: DMA a [rows x rowlen] bf16 tile (row stride `row_stride` elems) into LDS,
// packed row-major contiguously at lds_addr. D# packing per ISA 08_async_tensor §8.
static __device__ inline void tdm_load_tile_bf16(const __bf16* gsrc, unsigned lds_addr,
                                                 int rows, int rowlen, int row_stride) {
    unsigned long long ga = (unsigned long long)(uintptr_t)gsrc;
    v4u g0;
    g0[0] = 1u;                                          // count=1 (valid), user mode
    g0[1] = lds_addr;                                    // lds_addr [63:32]
    g0[2] = (unsigned)(ga & 0xFFFFFFFFu);                // global_addr lo
    g0[3] = (unsigned)((ga >> 32) & 0x01FFFFFFu)         // global_addr [56:32]
            | (2u << 30);                                // type=2 ("image")
    unsigned td0 = (unsigned)row_stride;                 // tensor_dim0 (elems)
    unsigned td1 = 1u << 20;                             // tensor_dim1: generous (no OOB)
    v8i_t g1;
    g1[0] = (int)(1u << 16);                             // data_size=1 -> 2 bytes
    g1[1] = (int)((td0 & 0xFFFFu) << 16);                // tensor_dim0 lo16 @ [63:48]
    g1[2] = (int)(((td0 >> 16) & 0xFFFFu)                // tensor_dim0 hi16 @ [79:64]
            | ((td1 & 0xFFFFu) << 16));                  // tensor_dim1 lo16 @ [95:80]
    g1[3] = (int)(((td1 >> 16) & 0xFFFFu)                // tensor_dim1 hi16 @ [111:96]
            | (((unsigned)rowlen & 0xFFFFu) << 16));     // tile_dim0 @ [127:112]
    g1[4] = (int)((unsigned)rows & 0xFFFFu);             // tile_dim1 @ [143:128], tile_dim2=0
    g1[5] = (int)((unsigned)row_stride);                 // tensor_dim0_stride lo32
    g1[6] = 0;                                           // stride hi + dim1_stride lo
    g1[7] = 0;
    v4i gz = {0, 0, 0, 0};
#if __clang_major__ >= 23
    v8i_t gz8 = {0, 0, 0, 0, 0, 0, 0, 0};
    __builtin_amdgcn_tensor_load_to_lds(g0, g1, gz, gz, gz8, 0);
#else
    __builtin_amdgcn_tensor_load_to_lds(g0, g1, gz, gz, 0);
#endif
}
#endif // USE_TDM

// ---------------------------------------------------------------------------
// Elementwise f32 -> bf16
// ---------------------------------------------------------------------------
__global__ void cvt_f32_bf16(const float* __restrict__ in, __bf16* __restrict__ out, int n) {
    int i = blockIdx.x * 256 + threadIdx.x;
    if (i < n) out[i] = (__bf16)in[i];
}

// ---------------------------------------------------------------------------
// C[m,n] = sum_k A[m,k] * W[n,k] + bias[n]   (A: MxK, W: NxK, both bf16)
// Block = 256 thr (8 waves), block tile 128(M) x 64(N), k-chunk 32.
// A(128x32) and W(64x32) tiles staged in LDS by the Tensor Data Mover,
// double-buffered: wave 0 issues chunk i+1's DMA, waits TENSORcnt<=2 (chunk i
// complete, chunk i+1 in flight), barrier, all waves consume LDS fragments.
// mode 0: bf16 out [b,h,s,d] | mode 2: bf16 out [b,h,d,s] | mode 1: f32 [m,n]
// ---------------------------------------------------------------------------
__global__ void gemm_abt(const __bf16* __restrict__ A, const __bf16* __restrict__ W,
                         const float* __restrict__ bias, float* __restrict__ outF,
                         __bf16* __restrict__ outB, int M, int N, int K, int mode, int S) {
    const int H = 16;
    __shared__ __bf16 Al[2][128 * 32];   // 16 KB
    __shared__ __bf16 Wl[2][64 * 32];    //  8 KB
    int lane = threadIdx.x & 31;
    int w    = threadIdx.x >> 5;
    int n0   = blockIdx.x * 64;
    int m0   = blockIdx.y * 128;
    int niter = K / 32;

#if USE_TDM
    if (w == 0) {  // prologue: chunk 0 in flight
        tdm_load_tile_bf16(A + (size_t)m0 * K, lds_off(&Al[0][0]), 128, 32, K);
        tdm_load_tile_bf16(W + (size_t)n0 * K, lds_off(&Wl[0][0]), 64, 32, K);
    }
#endif

    v8f c[4] = {};
    for (int it = 0; it < niter; ++it) {
        int kc  = it * 32;
        int buf = it & 1;
#if USE_TDM
        if (w == 0) {
            if (it + 1 < niter) {
                tdm_load_tile_bf16(A + (size_t)m0 * K + kc + 32,
                                   lds_off(&Al[buf ^ 1][0]), 128, 32, K);
                tdm_load_tile_bf16(W + (size_t)n0 * K + kc + 32,
                                   lds_off(&Wl[buf ^ 1][0]), 64, 32, K);
                __builtin_amdgcn_s_wait_tensorcnt(2);  // chunk `it` complete
            } else {
                __builtin_amdgcn_s_wait_tensorcnt(0);
            }
        }
        __syncthreads();
#else
        __syncthreads();
        // cooperative fallback copy: A tile (512 x v8bf), W tile (256 x v8bf)
        for (int i = threadIdx.x; i < 512; i += 256) {
            int r = i >> 2, seg = i & 3;
            ((v8bf*)&Al[buf][0])[i] =
                *(const v8bf*)(A + (size_t)(m0 + r) * K + kc + seg * 8);
        }
        for (int i = threadIdx.x; i < 256; i += 256) {
            int r = i >> 2, seg = i & 3;
            ((v8bf*)&Wl[buf][0])[i] =
                *(const v8bf*)(W + (size_t)(n0 + r) * K + kc + seg * 8);
        }
        __syncthreads();
#endif
        v16bf a = load_frag_a(&Al[buf][(w * 16) * 32], 32, lane);
#pragma unroll
        for (int g = 0; g < 4; ++g) {
            v16bf b = load_frag_bT(&Wl[buf][(g * 16) * 32], 32, lane);
            c[g] = wmma_bf16(a, b, c[g]);
        }
        __syncthreads();  // all waves done with `buf` before its DMA reuse
    }

#pragma unroll
    for (int g = 0; g < 4; ++g) {
#pragma unroll
        for (int v = 0; v < 8; ++v) {
            int m = m0 + w * 16 + ((lane < 16) ? v : v + 8);
            int n = n0 + g * 16 + (lane & 15);
            float val = c[g][v] + bias[n];
            if (mode == 1) {
                outF[(size_t)m * N + n] = val;
            } else {
                int b = m / S, s = m % S;
                int h = n >> 6, d = n & 63;
                if (mode == 0)
                    outB[(((size_t)b * H + h) * S + s) * 64 + d] = (__bf16)val;
                else
                    outB[(((size_t)b * H + h) * 64 + d) * S + s] = (__bf16)val;
            }
        }
    }
}

// ---------------------------------------------------------------------------
// scores[bh, q, k] = (Q[bh,q,:] . K[bh,k,:]) / 8
// grid = B*H*(S/16); block = 256; wave w covers key cols [w*256, w*256+256)
// ---------------------------------------------------------------------------
__global__ void scores_kernel(const __bf16* __restrict__ Qh, const __bf16* __restrict__ Kh,
                              float* __restrict__ attn, int S) {
    int nq   = S >> 4;
    int bh   = blockIdx.x / nq;
    int q0   = (blockIdx.x % nq) << 4;
    int lane = threadIdx.x & 31;
    int w    = threadIdx.x >> 5;

    const __bf16* Qb = Qh + ((size_t)bh * S + q0) * 64;
    v16bf aq0 = load_frag_a(Qb, 64, lane);        // d = 0..31
    v16bf aq1 = load_frag_a(Qb + 32, 64, lane);   // d = 32..63

    float* out = attn + ((size_t)bh * S + q0) * S;
    for (int n0 = w * 256; n0 < w * 256 + 256; n0 += 16) {
        const __bf16* Kb = Kh + ((size_t)bh * S + n0) * 64;
        __builtin_prefetch(Kb + 16 * 64, 0, 0);   // next K panel -> global_prefetch
        v16bf bk0 = load_frag_bT(Kb, 64, lane);
        v16bf bk1 = load_frag_bT(Kb + 32, 64, lane);
        v8f c = {};
        c = wmma_bf16(aq0, bk0, c);
        c = wmma_bf16(aq1, bk1, c);
#pragma unroll
        for (int v = 0; v < 8; ++v) {
            int r = (lane < 16) ? v : v + 8;
            out[(size_t)r * S + n0 + (lane & 15)] = c[v] * 0.125f;
        }
    }
}

// ---------------------------------------------------------------------------
// Row softmax in place over trailing S dim. grid = B*H*S rows.
// ---------------------------------------------------------------------------
__global__ void softmax_rows(float* __restrict__ attn, int S) {
    float* p = attn + (size_t)blockIdx.x * S;
    __shared__ float red[256];
    int t = threadIdx.x;

    float m = -1e30f;
    for (int i = t; i < S; i += 256) m = fmaxf(m, p[i]);
    red[t] = m; __syncthreads();
    for (int o = 128; o > 0; o >>= 1) { if (t < o) red[t] = fmaxf(red[t], red[t + o]); __syncthreads(); }
    m = red[0]; __syncthreads();

    float s = 0.f;
    for (int i = t; i < S; i += 256) { float e = __expf(p[i] - m); p[i] = e; s += e; }
    red[t] = s; __syncthreads();
    for (int o = 128; o > 0; o >>= 1) { if (t < o) red[t] += red[t + o]; __syncthreads(); }
    float inv = 1.0f / red[0];
    for (int i = t; i < S; i += 256) p[i] *= inv;
}

// ---------------------------------------------------------------------------
// X[b, q, h*64+d] = sum_k P[bh,q,k] * V[bh,k,d]   (V stored transposed [bh,d,s])
// grid = B*H*(S/16); block = 256; waves split keys; reduce via LDS ds_add_f32.
// ---------------------------------------------------------------------------
__global__ void pv_kernel(const float* __restrict__ attn, const __bf16* __restrict__ Vt,
                          __bf16* __restrict__ X, int S, int H) {
    int nq   = S >> 4;
    int bh   = blockIdx.x / nq;
    int q0   = (blockIdx.x % nq) << 4;
    int lane = threadIdx.x & 31;
    int w    = threadIdx.x >> 5;

    const float*  Ar = attn + ((size_t)bh * S + q0) * S;
    const __bf16* Vb = Vt + (size_t)bh * 64 * S;

    v8f c[4] = {};
    for (int kc = w * 32; kc < S; kc += 8 * 32) {
        v16bf a = load_frag_a_f32(Ar + kc, S, lane);
#pragma unroll
        for (int g = 0; g < 4; ++g) {
            v16bf b = load_frag_bT(Vb + (size_t)(g * 16) * S + kc, S, lane);
            c[g] = wmma_bf16(a, b, c[g]);
        }
    }

    __shared__ float acc[16 * 64];
    for (int i = threadIdx.x; i < 16 * 64; i += 256) acc[i] = 0.f;
    __syncthreads();
#pragma unroll
    for (int g = 0; g < 4; ++g)
#pragma unroll
        for (int v = 0; v < 8; ++v) {
            int r = (lane < 16) ? v : v + 8;
            int d = g * 16 + (lane & 15);
            atomicAdd(&acc[r * 64 + d], c[g][v]);
        }
    __syncthreads();

    int b = bh / H, h = bh % H;
    for (int i = threadIdx.x; i < 16 * 64; i += 256) {
        int r = i >> 6, d = i & 63;
        X[((size_t)b * S + q0 + r) * (H * 64) + h * 64 + d] = (__bf16)acc[i];
    }
}

// ---------------------------------------------------------------------------
// Host launch
// ---------------------------------------------------------------------------
extern "C" void kernel_launch(void* const* d_in, const int* in_sizes, int n_in,
                              void* d_out, int out_size, void* d_ws, size_t ws_size,
                              hipStream_t stream) {
    const int B = 2, S = 2048, HID = 1024, H = 16;
    const int M = B * S;                 // 4096
    const int NE_ACT = B * S * HID;      // 4,194,304
    const int NE_W   = HID * HID;        // 1,048,576

    const float* query  = (const float*)d_in[0];
    const float* key_in = (const float*)d_in[1];
    const float* value  = (const float*)d_in[2];
    const float* Wq = (const float*)d_in[3];  const float* bq = (const float*)d_in[4];
    const float* Wk = (const float*)d_in[5];  const float* bk = (const float*)d_in[6];
    const float* Wv = (const float*)d_in[7];  const float* bv = (const float*)d_in[8];
    const float* Wo = (const float*)d_in[9];  const float* bo = (const float*)d_in[10];

    float* outF = (float*)d_out;                       // [B,S,HID]
    float* attn = (float*)d_out + (size_t)NE_ACT;      // [B,H,S,S]

    // workspace carve-up (bf16), ~67 MB total
    __bf16* p = (__bf16*)d_ws;
    __bf16* qb  = p; p += NE_ACT;
    __bf16* kb  = p; p += NE_ACT;
    __bf16* vb  = p; p += NE_ACT;
    __bf16* Wqb = p; p += NE_W;
    __bf16* Wkb = p; p += NE_W;
    __bf16* Wvb = p; p += NE_W;
    __bf16* Wob = p; p += NE_W;
    __bf16* Qh  = p; p += NE_ACT;   // [B,H,S,64]
    __bf16* Kh  = p; p += NE_ACT;   // [B,H,S,64]
    __bf16* Vt  = p; p += NE_ACT;   // [B,H,64,S]
    __bf16* Xc  = p; p += NE_ACT;   // [B,S,HID] context

    // 1) f32 -> bf16 conversions
    cvt_f32_bf16<<<(NE_ACT + 255) / 256, 256, 0, stream>>>(query,  qb,  NE_ACT);
    cvt_f32_bf16<<<(NE_ACT + 255) / 256, 256, 0, stream>>>(key_in, kb,  NE_ACT);
    cvt_f32_bf16<<<(NE_ACT + 255) / 256, 256, 0, stream>>>(value,  vb,  NE_ACT);
    cvt_f32_bf16<<<(NE_W   + 255) / 256, 256, 0, stream>>>(Wq,     Wqb, NE_W);
    cvt_f32_bf16<<<(NE_W   + 255) / 256, 256, 0, stream>>>(Wk,     Wkb, NE_W);
    cvt_f32_bf16<<<(NE_W   + 255) / 256, 256, 0, stream>>>(Wv,     Wvb, NE_W);
    cvt_f32_bf16<<<(NE_W   + 255) / 256, 256, 0, stream>>>(Wo,     Wob, NE_W);

    // 2) QKV projections (TDM-staged WMMA GEMMs); V stored pre-transposed
    dim3 ggrid(HID / 64, M / 128);
    gemm_abt<<<ggrid, 256, 0, stream>>>(qb, Wqb, bq, nullptr, Qh, M, HID, HID, 0, S);
    gemm_abt<<<ggrid, 256, 0, stream>>>(kb, Wkb, bk, nullptr, Kh, M, HID, HID, 0, S);
    gemm_abt<<<ggrid, 256, 0, stream>>>(vb, Wvb, bv, nullptr, Vt, M, HID, HID, 2, S);

    // 3) scores = Q K^T / 8  (straight into d_out's attn region)
    scores_kernel<<<B * H * (S / 16), 256, 0, stream>>>(Qh, Kh, attn, S);

    // 4) softmax rows in place
    softmax_rows<<<B * H * S, 256, 0, stream>>>(attn, S);

    // 5) context = P V  (attn f32 -> bf16 fragments on the fly)
    pv_kernel<<<B * H * (S / 16), 256, 0, stream>>>(attn, Vt, Xc, S, H);

    // 6) out = X Wo^T + bo (f32 output)
    gemm_abt<<<ggrid, 256, 0, stream>>>(Xc, Wob, bo, outF, nullptr, M, HID, HID, 1, S);
}
